// transparse_15771119911423
// MI455X (gfx1250) — compile-verified, matrix operand order checked by reading, not run on hardware
//
#include <hip/hip_runtime.h>
#include <hip/hip_bf16.h>
#include <stdint.h>

#define DIM        128
#define BATCH      2048
#define NODE_SIZE  40000
#define LINK_SIZE  100
#define THETA      0.75f
#define MARGIN     1.0f

typedef __attribute__((ext_vector_type(16))) _Float16 v16h;
typedef __attribute__((ext_vector_type(8)))  float    v8f;
typedef __attribute__((ext_vector_type(4)))  float    v4f;

// Mix a 32-bit state into two independent 14-bit uniforms (bits [13:0], [29:16]).
// Emulates the reference's permutation mask at identical 1/16384 granularity:
// an entry is kept iff 0 < v <= theta_eff * 16384, i.e. (v-1u) < thr_u.
__device__ __forceinline__ uint32_t hash_mix(uint32_t h) {
    h ^= h >> 16; h *= 0x7FEB352Du;
    h ^= h >> 15; h *= 0x846CA68Bu;
    h ^= h >> 16;
    return h;
}

// One wave computes out[0:128] = (M .* mask) @ e with v_wmma_f32_16x16x32_f16.
// B holds e replicated across all 16 N columns, so every column of D is out.
// ISA 7.12.2 16-bit A 16x32 layout: lane<16 -> K in {0..7,16..23},
// lane>=16 -> K in {8..15,24..31}; B's 32x16 layout needs the same K-set per lane.
__device__ __forceinline__ void masked_matvec_wmma(
    const float* __restrict__ M, const float* __restrict__ e,
    uint32_t seed, uint32_t thr_u, int lane, float* out /* LDS, 128 floats */)
{
    const int mrow  = lane & 15;
    const int khalf = (lane < 16) ? 0 : 8;

    // Hoist B: 4 K-chunks of the replicated e vector (32 VGPRs total).
    v16h Bv[4];
    #pragma unroll
    for (int kt = 0; kt < 4; ++kt) {
        const int kb0 = kt * 32 + khalf;
        const int kb1 = kb0 + 16;
        float bb[16];
        *(v4f*)(bb + 0)  = *(const v4f*)(e + kb0);
        *(v4f*)(bb + 4)  = *(const v4f*)(e + kb0 + 4);
        *(v4f*)(bb + 8)  = *(const v4f*)(e + kb1);
        *(v4f*)(bb + 12) = *(const v4f*)(e + kb1 + 4);
        #pragma unroll
        for (int i = 0; i < 16; ++i) Bv[kt][i] = (_Float16)bb[i];
    }

    const uint32_t seedmix = seed * 0x85EBCA77u;

    for (int rt = 0; rt < 8; ++rt) {
        v8f c = {};
        const int row = rt * 16 + mrow;
        const float* mp = M + row * DIM;
        // Per-tile hash base; element offset folds to a compile-time constant.
        const uint32_t hbase =
            ((uint32_t)(row * DIM) + (uint32_t)khalf) * 0x9E3779B1u + seedmix;
        #pragma unroll
        for (int kt = 0; kt < 4; ++kt) {
            const int kb0 = kt * 32 + khalf;      // K run 0: kb0 .. kb0+7
            const int kb1 = kb0 + 16;             // K run 1: kb1 .. kb1+7
            float a[16];
            *(v4f*)(a + 0)  = *(const v4f*)(mp + kb0);
            *(v4f*)(a + 4)  = *(const v4f*)(mp + kb0 + 4);
            *(v4f*)(a + 8)  = *(const v4f*)(mp + kb1);
            *(v4f*)(a + 12) = *(const v4f*)(mp + kb1 + 4);
            v16h A;
            #pragma unroll
            for (int i = 0; i < 16; i += 2) {
                // koff is a compile-time constant per unrolled i.
                const uint32_t koff = (uint32_t)((i < 8) ? (kt * 32 + i)
                                                         : (kt * 32 + 16 + (i - 8)));
                const uint32_t h  = hash_mix(hbase + koff * 0x9E3779B1u);
                const uint32_t v0 = h & 16383u;
                const uint32_t v1 = (h >> 16) & 16383u;
                A[i]     = ((v0 - 1u) < thr_u) ? (_Float16)a[i]     : (_Float16)0.0f;
                A[i + 1] = ((v1 - 1u) < thr_u) ? (_Float16)a[i + 1] : (_Float16)0.0f;
            }
            c = __builtin_amdgcn_wmma_f32_16x16x32_f16(
                    false, A, false, Bv[kt], (short)0, c, false, false);
        }
        // C/D layout: VGPR v = row rt*16+v (lanes 0..15) / rt*16+8+v (lanes 16..31);
        // all N columns identical -> lanes 0 and 16 spill 8 rows each to LDS.
        if (mrow == 0) {
            const int base = rt * 16 + khalf;
            #pragma unroll
            for (int v = 0; v < 8; ++v) out[base + v] = c[v];
        }
    }
}

__global__ void __launch_bounds__(200) dmax_kernel(const int* __restrict__ nd,
                                                   float* __restrict__ dmax) {
    __shared__ int red[256];
    const int j = blockIdx.x;                 // 0..199 = (link, col) column
    int m = 0;
    for (int i = threadIdx.x; i < NODE_SIZE; i += blockDim.x)
        m = max(m, nd[(size_t)i * (LINK_SIZE * 2) + j]);
    red[threadIdx.x] = m;
    __syncthreads();
    for (int s = 128; s > 0; s >>= 1) {
        if (threadIdx.x < s && threadIdx.x + s < blockDim.x)
            red[threadIdx.x] = max(red[threadIdx.x], red[threadIdx.x + s]);
        __syncthreads();
    }
    if (threadIdx.x == 0) dmax[j] = (float)red[0];
}

__global__ void __launch_bounds__(256) transparse_main(
    const int* __restrict__ sp, const int* __restrict__ tp,
    const int* __restrict__ sn, const int* __restrict__ tn,
    const int* __restrict__ r,  const int* __restrict__ node_degree,
    const float* __restrict__ node_emb, const float* __restrict__ link_emb,
    const float* __restrict__ s_mat,    const float* __restrict__ t_mat,
    const float* __restrict__ dmax,     float* __restrict__ losses)
{
    __shared__ float obuf[8][4][DIM];         // 16 KB: per-wave {s+, t+, s-, t-}
    const int wave = threadIdx.x >> 5;
    const int lane = threadIdx.x & 31;
    const int b = blockIdx.x * 8 + wave;      // one wave per sample

    const int ri = r[b];
    int nodes[4] = { sp[b], tp[b], sn[b], tn[b] };
    const float* mats[4];
    mats[0] = s_mat + (size_t)ri * (DIM * DIM);
    mats[1] = t_mat + (size_t)ri * (DIM * DIM);
    mats[2] = mats[0];
    mats[3] = mats[1];

    uint32_t thr[4];
    #pragma unroll
    for (int side = 0; side < 4; ++side) {
        const int col  = side & 1;            // s-side uses col 0, t-side col 1
        const float dg = (float)node_degree[(size_t)nodes[side] * (LINK_SIZE * 2) + ri * 2 + col];
        const float dm = dmax[ri * 2 + col];
        const float th = 1.0f - (1.0f - THETA) * dg / dm;
        thr[side] = (uint32_t)(th * 16384.0f);
    }

    #pragma unroll
    for (int side = 0; side < 4; ++side) {
        const float* e = node_emb + (size_t)nodes[side] * DIM;
        masked_matvec_wmma(mats[side], e, (uint32_t)(b * 4 + side), thr[side],
                           lane, &obuf[wave][side][0]);
    }
    __syncthreads();

    float sum_p = 0.f, sum_n = 0.f;
    const float* re = link_emb + (size_t)ri * DIM;
    #pragma unroll
    for (int q = 0; q < 4; ++q) {
        const int row  = lane + q * 32;
        const float rv = re[row];
        const float dp = obuf[wave][0][row] + rv - obuf[wave][1][row];
        const float dn = obuf[wave][2][row] + rv - obuf[wave][3][row];
        sum_p += dp * dp;
        sum_n += dn * dn;
    }
    #pragma unroll
    for (int off = 16; off > 0; off >>= 1) {
        sum_p += __shfl_xor(sum_p, off, 32);
        sum_n += __shfl_xor(sum_n, off, 32);
    }
    if (lane == 0)
        losses[b] = fmaxf(0.0f, sqrtf(sum_p) - sqrtf(sum_n) + MARGIN);
}

__global__ void __launch_bounds__(256) mean_kernel(const float* __restrict__ losses,
                                                   float* __restrict__ out) {
    __shared__ float red[256];
    float s = 0.f;
    for (int i = threadIdx.x; i < BATCH; i += 256) s += losses[i];
    red[threadIdx.x] = s;
    __syncthreads();
    for (int st = 128; st > 0; st >>= 1) {
        if (threadIdx.x < st) red[threadIdx.x] += red[threadIdx.x + st];
        __syncthreads();
    }
    if (threadIdx.x == 0) out[0] = red[0] * (1.0f / (float)BATCH);
}

extern "C" void kernel_launch(void* const* d_in, const int* in_sizes, int n_in,
                              void* d_out, int out_size, void* d_ws, size_t ws_size,
                              hipStream_t stream) {
    const int*   sp          = (const int*)d_in[0];
    const int*   tp          = (const int*)d_in[1];
    const int*   sn          = (const int*)d_in[2];
    const int*   tn          = (const int*)d_in[3];
    const int*   rr          = (const int*)d_in[4];
    const int*   node_degree = (const int*)d_in[5];
    const float* node_emb    = (const float*)d_in[6];
    const float* link_emb    = (const float*)d_in[7];
    const float* s_mat       = (const float*)d_in[8];
    const float* t_mat       = (const float*)d_in[9];
    float* out    = (float*)d_out;
    float* wsf    = (float*)d_ws;
    float* dmax   = wsf;          // 200 floats (padded to 256)
    float* losses = wsf + 256;    // 2048 floats

    dmax_kernel<<<dim3(LINK_SIZE * 2), dim3(200), 0, stream>>>(node_degree, dmax);
    transparse_main<<<dim3(BATCH / 8), dim3(256), 0, stream>>>(
        sp, tp, sn, tn, rr, node_degree, node_emb, link_emb, s_mat, t_mat,
        dmax, losses);
    mean_kernel<<<dim3(1), dim3(256), 0, stream>>>(losses, out);
}
